// GenerativeUpBlock_81484119539721
// MI455X (gfx1250) — compile-verified, matrix operand order checked by reading, not run on hardware
//
#include <hip/hip_runtime.h>
#include <stdint.h>

// ---------------------------------------------------------------------------
// GenerativeUpBlock for MI455X (gfx1250, wave32, WMMA)
//   out[Ng,32] = mask * ( sum_{off in 27} gather(in_feats, idx[off]) @ W[off] + bias )
// bf16 WMMA (v_wmma_f32_16x16x32_bf16), f32 accumulate.
// ---------------------------------------------------------------------------

typedef __bf16 bf16_t;
typedef bf16_t v16bf __attribute__((ext_vector_type(16)));
typedef float  v8f   __attribute__((ext_vector_type(8)));

#define N_IN   64
#define N_OUT  32
#define N_OFF  27

#define TBL_BITS 19u
#define TBL      (1u << TBL_BITS)
#define TBL_MASK (TBL - 1u)
#define EMPTY_KEY 0xFFFFFFFFFFFFFFFFull

__device__ __forceinline__ unsigned long long mix64(unsigned long long h) {
    h ^= h >> 33; h *= 0xff51afd7ed558ccdull;
    h ^= h >> 33; h *= 0xc4ceb9fe1a85ec53ull;
    h ^= h >> 33;
    return h;
}

__device__ __forceinline__ long long coord_key(long long b, long long x,
                                               long long y, long long z) {
    // same hash as reference: [1, 1e5, 1e10, 1e15]
    return b + x * 100000ll + y * 10000000000ll + z * 1000000000000000ll;
}

__device__ __forceinline__ unsigned short f2bf(float f) {
    unsigned int u = __float_as_uint(f);
    u += 0x7FFFu + ((u >> 16) & 1u);      // round-to-nearest-even
    return (unsigned short)(u >> 16);
}

// ---------------------------------------------------------------------------
// Kernel 1: clear hash table keys
// ---------------------------------------------------------------------------
__global__ void k_init(unsigned long long* __restrict__ hkeys) {
    unsigned i = blockIdx.x * blockDim.x + threadIdx.x;
    if (i < TBL) hkeys[i] = EMPTY_KEY;
}

// ---------------------------------------------------------------------------
// Kernel 2: pre-swizzle weights [27][64][32] f32 -> WMMA B-fragment bf16 layout
// wfrag[((off*4 + khalf*2 + nhalf)*32 + lane)*16 + e]
// lane<16: N=lane,    K = kb+{0..7, 16..23}, kb = khalf*32
// lane>=16:N=lane-16, K = kb+{8..15,24..31}
// ---------------------------------------------------------------------------
__global__ void k_wconv(const float* __restrict__ w, unsigned short* __restrict__ wf) {
    int off   = blockIdx.x;            // 27
    int r     = threadIdx.x;           // 128 = 2 khalf * 2 nhalf * 32 lanes
    int khalf = r >> 6;
    int r2    = r & 63;
    int nhalf = r2 >> 5;
    int lane  = r2 & 31;

    int kbase = khalf * 32 + ((lane >> 4) << 3);  // +0 or +8
    int n     = nhalf * 16 + (lane & 15);
    const float*    ws = w + off * (N_IN * N_OUT);
    unsigned short* d  = wf + (((off * 4 + khalf * 2 + nhalf) * 32 + lane) << 4);
#pragma unroll
    for (int e = 0; e < 16; ++e) {
        int k = kbase + (e < 8 ? e : 8 + e);      // e>=8 -> kbase+16+(e-8)
        d[e] = f2bf(ws[k * N_OUT + n]);
    }
}

// ---------------------------------------------------------------------------
// Kernel 3: insert input coords into open-addressing hash table
// ---------------------------------------------------------------------------
__global__ void k_insert(const long long* __restrict__ coords, int np,
                         unsigned long long* __restrict__ hkeys,
                         int* __restrict__ hvals) {
    int p = blockIdx.x * blockDim.x + threadIdx.x;
    if (p >= np) return;
    const long long* c = coords + (size_t)p * 4;
    unsigned long long key = (unsigned long long)coord_key(c[0], c[1], c[2], c[3]);
    unsigned slot = (unsigned)(mix64(key) & TBL_MASK);
    while (true) {
        unsigned long long old = atomicCAS(&hkeys[slot], EMPTY_KEY, key);
        if (old == EMPTY_KEY || old == key) { hvals[slot] = p; break; }
        slot = (slot + 1) & TBL_MASK;
    }
}

// ---------------------------------------------------------------------------
// Kernel 4: fused lookup + WMMA GEMM.
// One wave32 = one 16-row x 32-col output tile, 4 waves / block.
// ---------------------------------------------------------------------------
__global__ __launch_bounds__(128) void k_gemm(
    const float* __restrict__ in_feats,
    const float* __restrict__ bias,
    const long long* __restrict__ guide,
    const unsigned long long* __restrict__ hkeys,
    const int* __restrict__ hvals,
    const unsigned short* __restrict__ wfrag,
    float* __restrict__ out,
    int ng)
{
    __shared__ int sh_idx[4][N_OFF * 16];
    __shared__ int sh_hit[4][N_OFF];
    __shared__ int sh_ex [4][16];

    const int w     = threadIdx.x >> 5;
    const int lane  = threadIdx.x & 31;
    const int tile  = blockIdx.x * 4 + w;
    const int gbase = tile * 16;

    if (lane < N_OFF) sh_hit[w][lane] = 0;
    if (lane < 16)    sh_ex [w][lane] = 0;
    __syncthreads();

    // ---- lookup phase: 27 offsets x 16 rows distributed over the wave ----
    for (int t = lane; t < N_OFF * 16; t += 32) {
        int row = t & 15;
        int off = t >> 4;
        int idx = -1;
        int g   = gbase + row;
        if (g < ng) {
            const long long* c  = guide + (size_t)g * 4;
            long long dx = (long long)(off / 9) - 1;
            long long dy = (long long)((off / 3) % 3) - 1;
            long long dz = (long long)(off % 3) - 1;
            unsigned long long key =
                (unsigned long long)coord_key(c[0], c[1] - dx, c[2] - dy, c[3] - dz);
            unsigned slot = (unsigned)(mix64(key) & TBL_MASK);
            while (true) {
                unsigned long long k = hkeys[slot];
                if (k == EMPTY_KEY) break;
                if (k == key) { idx = hvals[slot]; break; }
                slot = (slot + 1) & TBL_MASK;
            }
        }
        sh_idx[w][t] = idx;
        if (idx >= 0) {
            atomicOr(&sh_hit[w][off], 1);
            atomicOr(&sh_ex [w][row], 1);
        }
    }
    __syncthreads();

    // ---- GEMM phase ----
    v8f c0 = {};   // output cols 0..15
    v8f c1 = {};   // output cols 16..31
    const int rsel  = (lane >> 4) << 3;   // 0 or 8 (K-chunk / C-row select)
    const int myrow = lane & 15;          // A-fragment row (M)

    for (int off = 0; off < N_OFF; ++off) {
        if (!sh_hit[w][off]) continue;    // uniform per wave -> EXEC stays all-ones
        const int ridx = sh_idx[w][off * 16 + myrow];
#pragma unroll
        for (int kh = 0; kh < 2; ++kh) {
            // A fragment: lane<16: M=lane, K=kb+{0..7,16..23}; lane>=16: K chunks +8
            union { v16bf v; unsigned short s[16]; } A;
            const int kb = kh * 32 + rsel;
            if (ridx >= 0) {
                const float* rp = in_feats + (size_t)ridx * N_IN + kb;
                float tmp[16];
                *(float4*)(tmp + 0)  = *(const float4*)(rp + 0);
                *(float4*)(tmp + 4)  = *(const float4*)(rp + 4);
                *(float4*)(tmp + 8)  = *(const float4*)(rp + 16);
                *(float4*)(tmp + 12) = *(const float4*)(rp + 20);
#pragma unroll
                for (int e = 0; e < 16; ++e) A.s[e] = f2bf(tmp[e]);
            } else {
#pragma unroll
                for (int e = 0; e < 16; ++e) A.s[e] = 0;
            }
            // B fragments: one contiguous 32B load each (pre-swizzled)
            union { v16bf v; uint4 q[2]; } B0, B1;
            const uint4* bp0 =
                (const uint4*)(wfrag + (((off * 4 + kh * 2 + 0) * 32 + lane) << 4));
            const uint4* bp1 =
                (const uint4*)(wfrag + (((off * 4 + kh * 2 + 1) * 32 + lane) << 4));
            B0.q[0] = bp0[0]; B0.q[1] = bp0[1];
            B1.q[0] = bp1[0]; B1.q[1] = bp1[1];

            c0 = __builtin_amdgcn_wmma_f32_16x16x32_bf16(
                     false, A.v, false, B0.v, (short)0, c0, false, false);
            c1 = __builtin_amdgcn_wmma_f32_16x16x32_bf16(
                     false, A.v, false, B1.v, (short)0, c1, false, false);
        }
    }

    // ---- epilogue: C/D layout: lane<16: M=j, N=lane; lane>=16: M=8+j, N=lane-16 ----
    const float b0 = bias[myrow];
    const float b1 = bias[myrow + 16];
#pragma unroll
    for (int j = 0; j < 8; ++j) {
        int row = rsel + j;
        int g   = gbase + row;
        if (g < ng) {
            float ex = sh_ex[w][row] ? 1.0f : 0.0f;
            out[(size_t)g * N_OUT + myrow]      = (c0[j] + b0) * ex;
            out[(size_t)g * N_OUT + myrow + 16] = (c1[j] + b1) * ex;
        }
    }
}

// ---------------------------------------------------------------------------
// Host launcher
// ---------------------------------------------------------------------------
extern "C" void kernel_launch(void* const* d_in, const int* in_sizes, int n_in,
                              void* d_out, int out_size, void* d_ws, size_t ws_size,
                              hipStream_t stream) {
    const float*     in_feats = (const float*)d_in[0];
    const float*     weights  = (const float*)d_in[1];
    const float*     bias     = (const float*)d_in[2];
    const long long* in_coord = (const long long*)d_in[3];
    const long long* guide    = (const long long*)d_in[4];
    float*           out      = (float*)d_out;

    const int np = in_sizes[3] / 4;
    const int ng = in_sizes[4] / 4;

    char* ws = (char*)d_ws;
    unsigned long long* hkeys = (unsigned long long*)ws;                 // TBL*8 B
    int*                hvals = (int*)(ws + (size_t)TBL * 8);            // TBL*4 B
    unsigned short*     wfrag = (unsigned short*)(ws + (size_t)TBL * 12);// 27*4*32*16*2 B

    hipLaunchKernelGGL(k_init, dim3((TBL + 255) / 256), dim3(256), 0, stream, hkeys);
    hipLaunchKernelGGL(k_wconv, dim3(N_OFF), dim3(128), 0, stream, weights, wfrag);
    hipLaunchKernelGGL(k_insert, dim3((np + 255) / 256), dim3(256), 0, stream,
                       in_coord, np, hkeys, hvals);

    const int ntiles  = (ng + 15) / 16;
    const int nblocks = (ntiles + 3) / 4;
    hipLaunchKernelGGL(k_gemm, dim3(nblocks), dim3(128), 0, stream,
                       in_feats, bias, guide, hkeys, hvals, wfrag, out, ng);
}
